// Seq2SeqBiDirectionalSearch_18442589570155
// MI455X (gfx1250) — matrix operand order repaired and checked by reading, not compiled, not run on hardware
//
#include <hip/hip_runtime.h>
#include <hip/hip_bf16.h>
#include <math.h>

// ---------------- types ----------------
typedef __attribute__((ext_vector_type(16))) __bf16 v16bf;
typedef __attribute__((ext_vector_type(8)))  __bf16 v8bf;
typedef __attribute__((ext_vector_type(8)))  float  v8f;

// ---------------- model dims ----------------
#define S_LEN 40
#define T_LEN 30
#define BATCH 64
#define EMB   620
#define HID   1000
#define ATT   1000
#define MXO   500
#define OUTV  30000

// padded dims (N padded to mult of 16, K padded to mult of 32)
#define G3P   3008   // 3*HID -> ld of gate buffers
#define HKP   1024   // HID as K
#define EKP   640    // EMB as K
#define A_NP  1008   // ATT / HID as N
#define CKP   2048   // 2*HID (enc_out) as K
#define XKP   2624   // 2*HID+EMB as K
#define TKP   3648   // 3*HID+EMB as K
#define OKP   512    // MXO as K

// =====================================================================
// WMMA GEMM: C(M,Np) = act( A(M,Kp)_bf16 @ W(Np,Kp)_bf16^T + bias )
// A row-major lda=Kp, W row-major ldw=Kp (classic x @ W.T).
// Each wave computes a 32 x (NT*16) block of C: 2 M-tiles x NT N-tiles,
// NT is a template constant so the inner loop is branch-free (EXEC stays
// all-ones through every v_wmma). B fragments are reused across the two
// M-tiles: 8 WMMAs per 12 b128 loads at NT=4.
// =====================================================================
template<int NT, int ACT, int BIAS>
__global__ __launch_bounds__(256)
void wmma_gemm(const __bf16* __restrict__ A, const __bf16* __restrict__ W,
               const float* __restrict__ bias, int Nvalid,
               float* __restrict__ C, int M, int Kp, int ldc,
               int nGroups, int ntileStart)
{
    const int lane    = threadIdx.x & 31;
    const int wave    = threadIdx.x >> 5;
    const int tilesM2 = M >> 5;                    // 32-row blocks
    const int wt      = blockIdx.x * 8 + wave;
    if (wt >= tilesM2 * nGroups) return;           // wave-uniform exit

    const int m0      = (wt % tilesM2) << 5;
    const int ntile0  = ntileStart + (wt / tilesM2) * NT;
    const int halfSel = lane >> 4;                 // 0 or 1
    const int colLane = lane & 15;

    const __bf16* Abase0 = A + (size_t)(m0 + colLane) * Kp + (halfSel << 4);
    const __bf16* Abase1 = Abase0 + (size_t)16 * Kp;
    const __bf16* Wbase  = W + (size_t)(ntile0 * 16 + colLane) * Kp + (halfSel << 4);

    v8f acc0[NT] = {};
    v8f acc1[NT] = {};
    for (int k0 = 0; k0 < Kp; k0 += 32) {
        union { v16bf v; v8bf h[2]; } a0, a1;
        a0.h[0] = *reinterpret_cast<const v8bf*>(Abase0 + k0);
        a0.h[1] = *reinterpret_cast<const v8bf*>(Abase0 + k0 + 8);
        a1.h[0] = *reinterpret_cast<const v8bf*>(Abase1 + k0);
        a1.h[1] = *reinterpret_cast<const v8bf*>(Abase1 + k0 + 8);
#pragma unroll
        for (int t = 0; t < NT; ++t) {
            v16bf bfr = *reinterpret_cast<const v16bf*>(Wbase + (size_t)t * 16 * Kp + k0);
            acc0[t] = __builtin_amdgcn_wmma_f32_16x16x32_bf16(
                false, a0.v, false, bfr, (short)0, acc0[t], false, false);
            acc1[t] = __builtin_amdgcn_wmma_f32_16x16x32_bf16(
                false, a1.v, false, bfr, (short)0, acc1[t], false, false);
        }
    }

    const int rowBase = m0 + (halfSel << 3);
#pragma unroll
    for (int t = 0; t < NT; ++t) {
        const int n = (ntile0 + t) * 16 + colLane;
        float bv = 0.f;
        if (BIAS && n < Nvalid) bv = bias[n];
#pragma unroll
        for (int i = 0; i < 8; ++i) {
            float v0 = acc0[t][i] + bv;
            float v1 = acc1[t][i] + bv;
            if (ACT == 1) { v0 = tanhf(v0); v1 = tanhf(v1); }
            C[(size_t)(rowBase + i) * ldc + n]      = v0;
            C[(size_t)(rowBase + 16 + i) * ldc + n] = v1;
        }
    }
}

// host-side launcher: full groups of 4 N-tiles + one tail launch
template<int ACT, int BIAS>
static void launch_gemm(const __bf16* A, const __bf16* W, const float* bias, int Nvalid,
                        float* C, int M, int Np, int Kp, int ldc, hipStream_t stream)
{
    const int tilesM2 = M / 32;
    const int tilesN  = Np / 16;
    const int full    = tilesN / 4;
    const int rem     = tilesN % 4;
    if (full) {
        int tasks = tilesM2 * full;
        wmma_gemm<4, ACT, BIAS><<<(tasks + 7) / 8, 256, 0, stream>>>(
            A, W, bias, Nvalid, C, M, Kp, ldc, full, 0);
    }
    if (rem) {
        int tasks = tilesM2;
        int blocks = (tasks + 7) / 8;
        if (rem == 1)
            wmma_gemm<1, ACT, BIAS><<<blocks, 256, 0, stream>>>(
                A, W, bias, Nvalid, C, M, Kp, ldc, 1, full * 4);
        else if (rem == 2)
            wmma_gemm<2, ACT, BIAS><<<blocks, 256, 0, stream>>>(
                A, W, bias, Nvalid, C, M, Kp, ldc, 1, full * 4);
        else
            wmma_gemm<3, ACT, BIAS><<<blocks, 256, 0, stream>>>(
                A, W, bias, Nvalid, C, M, Kp, ldc, 1, full * 4);
    }
}

// ---------------- f32 -> bf16 pad/convert (weights & activations) -----
__global__ __launch_bounds__(256)
void convert_pad(const float* __restrict__ src, __bf16* __restrict__ dst,
                 int Nvalid, int Kvalid, int Kp, int srcLd, int srcColOff)
{
    const int n = blockIdx.x;
    for (int k = threadIdx.x; k < Kp; k += blockDim.x) {
        float v = 0.f;
        if (n < Nvalid && k < Kvalid)
            v = src[(size_t)n * srcLd + srcColOff + k];
        dst[(size_t)n * Kp + k] = (__bf16)v;
    }
}

// ---------------- embedding gather -> padded bf16 rows ----------------
__global__ __launch_bounds__(256)
void gather_emb(const float* __restrict__ table, const int* __restrict__ idx,
                __bf16* __restrict__ dst, int E, int Kp)
{
    const int r = blockIdx.x;
    const float* srcRow = table + (size_t)idx[r] * E;
    for (int k = threadIdx.x; k < Kp; k += blockDim.x)
        dst[(size_t)r * Kp + k] = (__bf16)((k < E) ? srcRow[k] : 0.f);
}

// ---------------- GRU gate update -------------------------------------
__global__ __launch_bounds__(256)
void gru_update(const float* __restrict__ gx, const float* __restrict__ gh,
                const float* __restrict__ bx, const float* __restrict__ bh,
                const float* __restrict__ hprev, int hld,
                float* __restrict__ hnext, __bf16* __restrict__ hnext_bf,
                float* __restrict__ store, int store_ld)
{
    const int b = blockIdx.x;
    const float* gxr = gx + (size_t)b * G3P;
    const float* ghr = gh + (size_t)b * G3P;
    for (int j = threadIdx.x; j < HKP; j += blockDim.x) {
        if (j < HID) {
            float r  = 1.f / (1.f + expf(-(gxr[j] + bx[j] + ghr[j] + bh[j])));
            float z  = 1.f / (1.f + expf(-(gxr[HID+j] + bx[HID+j] + ghr[HID+j] + bh[HID+j])));
            float nn = tanhf(gxr[2*HID+j] + bx[2*HID+j] + r * (ghr[2*HID+j] + bh[2*HID+j]));
            float h2 = (1.f - z) * nn + z * hprev[(size_t)b * hld + j];
            hnext[(size_t)b * hld + j] = h2;
            hnext_bf[b * HKP + j] = (__bf16)h2;
            if (store) store[(size_t)b * store_ld + j] = h2;
        } else {
            hnext_bf[b * HKP + j] = (__bf16)0.f;
        }
    }
}

// ---------------- attention scores: one block per (s,b) ---------------
__global__ __launch_bounds__(256)
void attn_scores(const float* __restrict__ hW, const float* __restrict__ e_part,
                 const float* __restrict__ ab, const float* __restrict__ av,
                 float* __restrict__ scores)
{
    const int s = blockIdx.x, b = blockIdx.y;
    const float* ep = e_part + (size_t)(s * BATCH + b) * A_NP;
    const float* hw = hW + (size_t)b * A_NP;
    float acc = 0.f;
    for (int a = threadIdx.x; a < ATT; a += blockDim.x)
        acc += tanhf(hw[a] + ep[a] + ab[a]) * av[a];
    __shared__ float red[256];
    red[threadIdx.x] = acc; __syncthreads();
    for (int o = 128; o > 0; o >>= 1) {
        if (threadIdx.x < o) red[threadIdx.x] += red[threadIdx.x + o];
        __syncthreads();
    }
    if (threadIdx.x == 0) scores[s * BATCH + b] = red[0];
}

// ---------------- softmax over s + context ----------------------------
__global__ __launch_bounds__(256)
void softmax_ctx(const float* __restrict__ scores, const float* __restrict__ enc_out,
                 float* __restrict__ ctx)
{
    const int b = blockIdx.x;
    __shared__ float a[S_LEN];
    if (threadIdx.x == 0) {
        float mx = -3.402823e38f;
        for (int s = 0; s < S_LEN; ++s) { float v = scores[s * BATCH + b]; mx = v > mx ? v : mx; }
        float sum = 0.f;
        for (int s = 0; s < S_LEN; ++s) { float e = expf(scores[s * BATCH + b] - mx); a[s] = e; sum += e; }
        float inv = 1.f / sum;
        for (int s = 0; s < S_LEN; ++s) a[s] *= inv;
    }
    __syncthreads();
    for (int h = threadIdx.x; h < 2 * HID; h += blockDim.x) {
        float acc = 0.f;
        for (int s = 0; s < S_LEN; ++s)
            acc += a[s] * enc_out[((size_t)s * BATCH + b) * (2 * HID) + h];
        ctx[(size_t)b * (2 * HID) + h] = acc;
    }
}

// ---------------- build decoder GRU input x = [emb, ctx] --------------
__global__ __launch_bounds__(256)
void build_x(const __bf16* __restrict__ embt, const float* __restrict__ ctx,
             __bf16* __restrict__ x)
{
    const int b = blockIdx.x;
    for (int k = threadIdx.x; k < XKP; k += blockDim.x) {
        float v;
        if (k < EMB)            v = (float)embt[b * EKP + k];
        else if (k < EMB+2*HID) v = ctx[(size_t)b * (2*HID) + (k - EMB)];
        else                    v = 0.f;
        x[(size_t)b * XKP + k] = (__bf16)v;
    }
}

// ---------------- build maxout input t = [h2, ctx, emb] ---------------
__global__ __launch_bounds__(256)
void build_t(const float* __restrict__ h2, int hld, const float* __restrict__ ctx,
             const __bf16* __restrict__ embt, __bf16* __restrict__ t)
{
    const int b = blockIdx.x;
    for (int k = threadIdx.x; k < TKP; k += blockDim.x) {
        float v;
        if (k < HID)              v = h2[(size_t)b * hld + k];
        else if (k < 3*HID)       v = ctx[(size_t)b * (2*HID) + (k - HID)];
        else if (k < 3*HID+EMB)   v = (float)embt[b * EKP + (k - 3*HID)];
        else                      v = 0.f;
        t[(size_t)b * TKP + k] = (__bf16)v;
    }
}

// ---------------- maxout over pairs -----------------------------------
__global__ __launch_bounds__(256)
void maxout_pairs(const float* __restrict__ mx, __bf16* __restrict__ t2)
{
    const int b = blockIdx.x;
    for (int j = threadIdx.x; j < OKP; j += blockDim.x) {
        float v = 0.f;
        if (j < MXO) v = fmaxf(mx[(size_t)b * A_NP + 2*j], mx[(size_t)b * A_NP + 2*j + 1]);
        t2[(size_t)b * OKP + j] = (__bf16)v;
    }
}

// ---------------- argmax over vocab -----------------------------------
__global__ __launch_bounds__(256)
void argmax_row(const float* __restrict__ pred, int* __restrict__ inp)
{
    const int b = blockIdx.x;
    const float* p = pred + (size_t)b * OUTV;
    float best = -3.402823e38f; int bi = 0x7fffffff;
    for (int i = threadIdx.x; i < OUTV; i += blockDim.x) {
        float v = p[i];
        if (v > best || (v == best && i < bi)) { best = v; bi = i; }
    }
    __shared__ float sv[256]; __shared__ int si[256];
    sv[threadIdx.x] = best; si[threadIdx.x] = bi; __syncthreads();
    for (int o = 128; o > 0; o >>= 1) {
        if (threadIdx.x < o) {
            float v2 = sv[threadIdx.x + o]; int i2 = si[threadIdx.x + o];
            if (v2 > sv[threadIdx.x] || (v2 == sv[threadIdx.x] && i2 < si[threadIdx.x])) {
                sv[threadIdx.x] = v2; si[threadIdx.x] = i2;
            }
        }
        __syncthreads();
    }
    if (threadIdx.x == 0) inp[b] = si[0];
}

// ---------------- misc ------------------------------------------------
__global__ void fill_f32(float* __restrict__ p, float v, size_t n)
{
    size_t i = (size_t)blockIdx.x * blockDim.x + threadIdx.x;
    size_t stride = (size_t)gridDim.x * blockDim.x;
    for (; i < n; i += stride) p[i] = v;
}
__global__ void init_inp(const int* __restrict__ trg, int* __restrict__ inp)
{
    if (threadIdx.x < BATCH) inp[threadIdx.x] = trg[threadIdx.x];
}

// =====================================================================
extern "C" void kernel_launch(void* const* d_in, const int* in_sizes, int n_in,
                              void* d_out, int out_size, void* d_ws, size_t ws_size,
                              hipStream_t stream)
{
    const int*   src     = (const int*)  d_in[0];
    const int*   trg     = (const int*)  d_in[1];
    const float* enc_emb = (const float*)d_in[2];
    const float* wxf     = (const float*)d_in[3];
    const float* whf     = (const float*)d_in[4];
    const float* bxf     = (const float*)d_in[5];
    const float* bhf     = (const float*)d_in[6];
    const float* wxb     = (const float*)d_in[7];
    const float* whb     = (const float*)d_in[8];
    const float* bxb     = (const float*)d_in[9];
    const float* bhb     = (const float*)d_in[10];
    const float* fcw     = (const float*)d_in[11];
    const float* fcb     = (const float*)d_in[12];
    const float* attnw   = (const float*)d_in[13];
    const float* attnb   = (const float*)d_in[14];
    const float* attnv   = (const float*)d_in[15];
    const float* dec_emb = (const float*)d_in[16];
    const float* dwx     = (const float*)d_in[17];
    const float* dwh     = (const float*)d_in[18];
    const float* dbx     = (const float*)d_in[19];
    const float* dbh     = (const float*)d_in[20];
    const float* mxw     = (const float*)d_in[21];
    const float* mxb     = (const float*)d_in[22];
    const float* outw    = (const float*)d_in[23];
    const float* outb    = (const float*)d_in[24];
    float* out = (float*)d_out;

    // bump allocator over workspace
    char* base = (char*)d_ws; size_t off = 0;
    auto alloc = [&](size_t bytes) -> void* {
        void* p = base + off; off = (off + bytes + 255) & ~(size_t)255; return p;
    };

    // ---- bf16 weight buffers ----
    __bf16* wxf_b  = (__bf16*)alloc((size_t)G3P  * EKP * 2);
    __bf16* whf_b  = (__bf16*)alloc((size_t)G3P  * HKP * 2);
    __bf16* wxb_b  = (__bf16*)alloc((size_t)G3P  * EKP * 2);
    __bf16* whb_b  = (__bf16*)alloc((size_t)G3P  * HKP * 2);
    __bf16* fc_b   = (__bf16*)alloc((size_t)A_NP * HKP * 2);
    __bf16* awh_b  = (__bf16*)alloc((size_t)A_NP * HKP * 2);
    __bf16* awe_b  = (__bf16*)alloc((size_t)A_NP * CKP * 2);
    __bf16* dwx_b  = (__bf16*)alloc((size_t)G3P  * XKP * 2);
    __bf16* dwh_b  = (__bf16*)alloc((size_t)G3P  * HKP * 2);
    __bf16* mxw_b  = (__bf16*)alloc((size_t)A_NP * TKP * 2);
    __bf16* outw_b = (__bf16*)alloc((size_t)OUTV * OKP * 2);

    // ---- activation buffers ----
    __bf16* emb_bf     = (__bf16*)alloc((size_t)S_LEN*BATCH * EKP * 2);
    float*  gxF        = (float*) alloc((size_t)S_LEN*BATCH * G3P * 4);
    float*  gxB        = (float*) alloc((size_t)S_LEN*BATCH * G3P * 4);
    float*  enc_out    = (float*) alloc((size_t)S_LEN*BATCH * 2*HID * 4);
    __bf16* enc_out_bf = (__bf16*)alloc((size_t)S_LEN*BATCH * CKP * 2);
    float*  e_part     = (float*) alloc((size_t)S_LEN*BATCH * A_NP * 4);
    float*  ghBuf      = (float*) alloc((size_t)BATCH * G3P * 4);
    float*  h0f        = (float*) alloc((size_t)BATCH * A_NP * 4);
    float*  h1f        = (float*) alloc((size_t)BATCH * A_NP * 4);
    __bf16* h0b        = (__bf16*)alloc((size_t)BATCH * HKP * 2);
    __bf16* h1b        = (__bf16*)alloc((size_t)BATCH * HKP * 2);
    float*  hidF       = (float*) alloc((size_t)BATCH * A_NP * 4);
    __bf16* hidB       = (__bf16*)alloc((size_t)BATCH * HKP * 2);
    float*  hD1        = (float*) alloc((size_t)BATCH * A_NP * 4);
    __bf16* hD1b       = (__bf16*)alloc((size_t)BATCH * HKP * 2);
    float*  hW         = (float*) alloc((size_t)BATCH * A_NP * 4);
    float*  scores     = (float*) alloc((size_t)S_LEN*BATCH * 4);
    float*  ctx        = (float*) alloc((size_t)BATCH * 2*HID * 4);
    __bf16* dembB      = (__bf16*)alloc((size_t)BATCH * EKP * 2);
    __bf16* xB         = (__bf16*)alloc((size_t)BATCH * XKP * 2);
    __bf16* tB         = (__bf16*)alloc((size_t)BATCH * TKP * 2);
    float*  gxd        = (float*) alloc((size_t)BATCH * G3P * 4);
    float*  ghd        = (float*) alloc((size_t)BATCH * G3P * 4);
    float*  mxo        = (float*) alloc((size_t)BATCH * A_NP * 4);
    __bf16* t2B        = (__bf16*)alloc((size_t)BATCH * OKP * 2);
    int*    inp        = (int*)   alloc((size_t)BATCH * 4);
    (void)ws_size; (void)n_in; (void)in_sizes; (void)out_size;

    // ---- weight conversion (once per launch) ----
    convert_pad<<<G3P,  256, 0, stream>>>(wxf, wxf_b, 3*HID, EMB,     EKP, EMB,     0);
    convert_pad<<<G3P,  256, 0, stream>>>(whf, whf_b, 3*HID, HID,     HKP, HID,     0);
    convert_pad<<<G3P,  256, 0, stream>>>(wxb, wxb_b, 3*HID, EMB,     EKP, EMB,     0);
    convert_pad<<<G3P,  256, 0, stream>>>(whb, whb_b, 3*HID, HID,     HKP, HID,     0);
    convert_pad<<<A_NP, 256, 0, stream>>>(fcw, fc_b,  HID,   HID,     HKP, HID,     0);
    convert_pad<<<A_NP, 256, 0, stream>>>(attnw, awh_b, ATT, HID,     HKP, 3*HID,   0);
    convert_pad<<<A_NP, 256, 0, stream>>>(attnw, awe_b, ATT, 2*HID,   CKP, 3*HID,   HID);
    convert_pad<<<G3P,  256, 0, stream>>>(dwx, dwx_b, 3*HID, 2*HID+EMB, XKP, 2*HID+EMB, 0);
    convert_pad<<<G3P,  256, 0, stream>>>(dwh, dwh_b, 3*HID, HID,     HKP, HID,     0);
    convert_pad<<<A_NP, 256, 0, stream>>>(mxw, mxw_b, 2*MXO, 3*HID+EMB, TKP, 3*HID+EMB, 0);
    convert_pad<<<OUTV, 256, 0, stream>>>(outw, outw_b, OUTV, MXO,    OKP, MXO,     0);

    // ================= ENCODER =================
    gather_emb<<<S_LEN*BATCH, 256, 0, stream>>>(enc_emb, src, emb_bf, EMB, EKP);

    // batched gx for both directions: (2560 x 640) @ (3008 x 640)^T
    launch_gemm<0,0>(emb_bf, wxf_b, nullptr, 0, gxF, S_LEN*BATCH, G3P, EKP, G3P, stream);
    launch_gemm<0,0>(emb_bf, wxb_b, nullptr, 0, gxB, S_LEN*BATCH, G3P, EKP, G3P, stream);

    // forward scan
    {
        float* hc = h0f; float* hn = h1f; __bf16* hcb = h0b; __bf16* hnb = h1b;
        fill_f32<<<256, 256, 0, stream>>>(hc, 0.f, (size_t)BATCH*A_NP);
        fill_f32<<<256, 256, 0, stream>>>((float*)hcb, 0.f, (size_t)BATCH*HKP/2);
        for (int t = 0; t < S_LEN; ++t) {
            launch_gemm<0,0>(hcb, whf_b, nullptr, 0, ghBuf, BATCH, G3P, HKP, G3P, stream);
            gru_update<<<BATCH, 256, 0, stream>>>(
                gxF + (size_t)t*BATCH*G3P, ghBuf, bxf, bhf,
                hc, A_NP, hn, hnb, enc_out + (size_t)t*BATCH*2*HID, 2*HID);
            float* tf = hc; hc = hn; hn = tf;
            __bf16* tb = hcb; hcb = hnb; hnb = tb;
        }
    }
    // backward scan (writes enc_out cols [HID,2*HID) at source position s)
    __bf16* hbackB;
    {
        float* hc = h0f; float* hn = h1f; __bf16* hcb = h0b; __bf16* hnb = h1b;
        fill_f32<<<256, 256, 0, stream>>>(hc, 0.f, (size_t)BATCH*A_NP);
        fill_f32<<<256, 256, 0, stream>>>((float*)hcb, 0.f, (size_t)BATCH*HKP/2);
        for (int t = 0; t < S_LEN; ++t) {
            int s = S_LEN - 1 - t;
            launch_gemm<0,0>(hcb, whb_b, nullptr, 0, ghBuf, BATCH, G3P, HKP, G3P, stream);
            gru_update<<<BATCH, 256, 0, stream>>>(
                gxB + (size_t)s*BATCH*G3P, ghBuf, bxb, bhb,
                hc, A_NP, hn, hnb, enc_out + (size_t)s*BATCH*2*HID + HID, 2*HID);
            float* tf = hc; hc = hn; hn = tf;
            __bf16* tb = hcb; hcb = hnb; hnb = tb;
        }
        hbackB = hcb;  // final backward hidden (bf16, padded)
    }

    // enc_out -> bf16 (padded K=2048) for e_part GEMM
    convert_pad<<<S_LEN*BATCH, 256, 0, stream>>>(enc_out, enc_out_bf,
        S_LEN*BATCH, 2*HID, CKP, 2*HID, 0);
    // e_part = enc_out @ aw_e^T : (2560 x 2048) @ (1008 x 2048)^T
    launch_gemm<0,0>(enc_out_bf, awe_b, nullptr, 0, e_part, S_LEN*BATCH, A_NP, CKP, A_NP, stream);

    // hidden = tanh(h_b @ fc^T + fc_b)
    launch_gemm<1,1>(hbackB, fc_b, fcb, HID, hidF, BATCH, A_NP, HKP, A_NP, stream);
    convert_pad<<<BATCH, 256, 0, stream>>>(hidF, hidB, BATCH, HID, HKP, A_NP, 0);

    // ================= DECODER =================
    init_inp<<<1, 64, 0, stream>>>(trg, inp);
    fill_f32<<<2048, 256, 0, stream>>>(out, 0.f, (size_t)BATCH*OUTV);  // row 0 zeros

    float* dc = hidF; __bf16* dcb = hidB;
    float* dn = hD1;  __bf16* dnb = hD1b;
    for (int step = 0; step < T_LEN - 1; ++step) {
        // attention
        launch_gemm<0,0>(dcb, awh_b, nullptr, 0, hW, BATCH, A_NP, HKP, A_NP, stream);
        attn_scores<<<dim3(S_LEN, BATCH), 256, 0, stream>>>(hW, e_part, attnb, attnv, scores);
        softmax_ctx<<<BATCH, 256, 0, stream>>>(scores, enc_out, ctx);
        // GRU input
        gather_emb<<<BATCH, 256, 0, stream>>>(dec_emb, inp, dembB, EMB, EKP);
        build_x<<<BATCH, 256, 0, stream>>>(dembB, ctx, xB);
        launch_gemm<0,0>(xB,  dwx_b, nullptr, 0, gxd, BATCH, G3P, XKP, G3P, stream);
        launch_gemm<0,0>(dcb, dwh_b, nullptr, 0, ghd, BATCH, G3P, HKP, G3P, stream);
        gru_update<<<BATCH, 256, 0, stream>>>(gxd, ghd, dbx, dbh,
            dc, A_NP, dn, dnb, nullptr, 0);
        // maxout + vocab projection
        build_t<<<BATCH, 256, 0, stream>>>(dn, A_NP, ctx, dembB, tB);
        launch_gemm<0,1>(tB, mxw_b, mxb, 2*MXO, mxo, BATCH, A_NP, TKP, A_NP, stream);
        maxout_pairs<<<BATCH, 256, 0, stream>>>(mxo, t2B);
        float* predRow = out + (size_t)(step + 1) * BATCH * OUTV;
        launch_gemm<0,1>(t2B, outw_b, outb, OUTV, predRow, BATCH, OUTV, OKP, OUTV, stream);
        argmax_row<<<BATCH, 256, 0, stream>>>(predRow, inp);
        // swap hidden ping-pong
        float* tf = dc; dc = dn; dn = tf;
        __bf16* tb = dcb; dcb = dnb; dnb = tb;
    }
}
